// AudioVisualModel_764504179027
// MI455X (gfx1250) — compile-verified, matrix operand order checked by reading, not run on hardware
//
#include <hip/hip_runtime.h>

// ---------------------------------------------------------------------------
// Types for CDNA5 WMMA (wave32): V_WMMA_F32_16X16X32_BF16
// ---------------------------------------------------------------------------
typedef __attribute__((ext_vector_type(16))) __bf16   v16bf;
typedef __attribute__((ext_vector_type(8)))  float    v8f;
typedef __attribute__((ext_vector_type(4)))  unsigned u32x4;

struct Frag32B { u32x4 lo, hi; };  // 16 bf16 = 32 bytes

// f32 -> bf16 round-to-nearest-even
__device__ __forceinline__ unsigned short f2bf(float f) {
  union { float f; unsigned u; } c; c.f = f;
  unsigned u = c.u;
  u = (u + 0x7FFFu + ((u >> 16) & 1u)) >> 16;
  return (unsigned short)u;
}

// ---------------------------------------------------------------------------
// Kernel 1: L2-normalize rows of (b, src_rows, 1024) f32 -> (b, dst_rows, 1024)
// bf16, zero-padding rows in [src_rows, dst_rows).
// ---------------------------------------------------------------------------
__global__ __launch_bounds__(256)
void norm_rows_kernel(const float* __restrict__ src,
                      unsigned short* __restrict__ dst,
                      int src_rows, int dst_rows) {
  const int D = 1024;
  int gr = blockIdx.x;
  int b  = gr / dst_rows;
  int r  = gr - b * dst_rows;
  int tid = threadIdx.x;
  unsigned short* drow = dst + ((size_t)b * dst_rows + r) * D;

  if (r >= src_rows) {                      // zero pad rows
    for (int k = tid; k < D; k += 256) drow[k] = 0;
    return;
  }
  const float* srow = src + ((size_t)b * src_rows + r) * D;

  float vals[4];
  float s = 0.0f;
  #pragma unroll
  for (int k = 0; k < 4; ++k) {
    vals[k] = srow[tid + 256 * k];
    s += vals[k] * vals[k];
  }
  __shared__ float red[256];
  red[tid] = s;
  __syncthreads();
  for (int off = 128; off > 0; off >>= 1) {
    if (tid < off) red[tid] += red[tid + off];
    __syncthreads();
  }
  float inv = 1.0f / fmaxf(sqrtf(red[0]), 1e-12f);
  #pragma unroll
  for (int k = 0; k < 4; ++k) drow[tid + 256 * k] = f2bf(vals[k] * inv);
}

// ---------------------------------------------------------------------------
// Kernel 2: fused all-pairs GEMM + (max over Nv, mean over Na) + nonneg reg.
// One block per (i,j) pair; 8 wave32s; bf16 WMMA, f32 accumulate.
//   A panel (64 x 1024 bf16 = 128 KB) staged in LDS (WGP has 320 KB).
//   B (v_j rows) streamed from global; all of v (64 MB bf16) is L2-resident.
// ---------------------------------------------------------------------------
__global__ __launch_bounds__(256)
void av_gemm_fused(const unsigned short* __restrict__ aB,   // [32][64][1024] bf16
                   const unsigned short* __restrict__ vB,   // [32][1024][1024] bf16
                   float* __restrict__ clip,                // [32][32]
                   float* __restrict__ nn_acc,              // scalar accumulator
                   const float* __restrict__ tptr) {
  __shared__ u32x4 a_sh[8192];          // 64 rows x 1024 bf16 = 128 KB
  __shared__ float pmax[8][16];         // per-wave row maxima
  __shared__ float rowmax[64];
  __shared__ float s_nn;

  const int i = blockIdx.x >> 5;
  const int j = blockIdx.x & 31;
  const int tid = threadIdx.x;

  // Stage audio panel i into LDS (coalesced 16B chunks)
  const u32x4* asrc = reinterpret_cast<const u32x4*>(aB) + (size_t)i * 8192;
  for (int t = tid; t < 8192; t += 256) a_sh[t] = asrc[t];
  if (tid == 0) s_nn = 0.0f;
  __syncthreads();

  const int wave = tid >> 5;
  const int lane = tid & 31;
  const int rt   = wave >> 1;           // audio row tile 0..3
  const int hf   = lane >> 4;           // lane half (K-group select)
  const int n    = lane & 15;           // M (A) / N (B) index within tile

  // A fragment source: row (rt*16+n), 16B-granular K chunks
  const u32x4* abase = a_sh + (size_t)(rt * 16 + n) * 128 + hf;
  const u32x4* vbatch = reinterpret_cast<const u32x4*>(vB) + (size_t)j * 131072;

  float rmax[8];
  #pragma unroll
  for (int r = 0; r < 8; ++r) rmax[r] = -1e30f;
  float nn_part = 0.0f;

  const int ct0 = (wave & 1) * 32;      // 32 visual col-tiles per wave
  for (int ct = ct0; ct < ct0 + 32; ++ct) {
    const u32x4* vbase = vbatch + (size_t)(ct * 16 + n) * 128 + hf;
    v8f c = {};
    #pragma unroll 4
    for (int k8 = 0; k8 < 128; k8 += 4) {       // K = 1024 in steps of 32
      __builtin_prefetch((const void*)(vbase + k8 + 16), 0, 1);
      Frag32B fa { abase[k8], abase[k8 + 2] };
      Frag32B fb { vbase[k8], vbase[k8 + 2] };
      c = __builtin_amdgcn_wmma_f32_16x16x32_bf16(
              false, __builtin_bit_cast(v16bf, fa),
              false, __builtin_bit_cast(v16bf, fb),
              (short)0, c, false, false);
    }
    // Epilogue: raw dot s (padded audio rows give exactly 0 -> no effect)
    #pragma unroll
    for (int r = 0; r < 8; ++r) {
      float s = c[r];
      rmax[r] = fmaxf(rmax[r], s);
      float m = fminf(s, 0.0f);
      nn_part += m * m;
    }
  }

  // Reduce row-max across the 16 lanes of each half (xor stays inside half)
  #pragma unroll
  for (int r = 0; r < 8; ++r) {
    float v = rmax[r];
    v = fmaxf(v, __shfl_xor(v, 1, 32));
    v = fmaxf(v, __shfl_xor(v, 2, 32));
    v = fmaxf(v, __shfl_xor(v, 4, 32));
    v = fmaxf(v, __shfl_xor(v, 8, 32));
    rmax[r] = v;
  }
  if (n == 0) {
    #pragma unroll
    for (int r = 0; r < 8; ++r) pmax[wave][hf * 8 + r] = rmax[r];
  }
  // Reduce nonneg partial across the wave
  nn_part += __shfl_xor(nn_part, 16, 32);
  nn_part += __shfl_xor(nn_part,  8, 32);
  nn_part += __shfl_xor(nn_part,  4, 32);
  nn_part += __shfl_xor(nn_part,  2, 32);
  nn_part += __shfl_xor(nn_part,  1, 32);
  if (lane == 0) atomicAdd(&s_nn, nn_part);
  __syncthreads();

  if (tid < 64) {
    int wr = (tid >> 4) * 2;
    rowmax[tid] = fmaxf(pmax[wr][tid & 15], pmax[wr + 1][tid & 15]);
  }
  __syncthreads();

  if (tid == 0) {
    float T = tptr[0];
    float sum = 0.0f;
    for (int R = 0; R < 50; ++R) sum += rowmax[R];   // only valid audio tokens
    clip[i * 32 + j] = sum / (50.0f * T);            // mean(max(sims))/temp
    atomicAdd(nn_acc, s_nn);                         // raw (un-scaled) sum
  }
}

// ---------------------------------------------------------------------------
// Kernel 3: InfoNCE over 32x32 clip sims + regularizers -> scalar loss.
// ---------------------------------------------------------------------------
__global__ void finalize_kernel(const float* __restrict__ clip,
                                const float* __restrict__ nn_acc,
                                const float* __restrict__ tptr,
                                float* __restrict__ out) {
  __shared__ float sc[32][33];
  __shared__ float losses[32];
  const int t = threadIdx.x;
  const float inv_sqrtB = 0.17677669529663687f;   // 1/sqrt(32)

  for (int jj = 0; jj < 32; ++jj) sc[t][jj] = clip[t * 32 + jj] * inv_sqrtB;
  __syncthreads();

  float m = -1e30f;
  for (int jj = 0; jj < 32; ++jj) m = fmaxf(m, sc[t][jj]);
  float se = 0.0f;
  for (int jj = 0; jj < 32; ++jj) se += expf(sc[t][jj] - m);
  float lse_row = m + logf(se);

  float m2 = -1e30f;
  for (int jj = 0; jj < 32; ++jj) m2 = fmaxf(m2, sc[jj][t]);
  float se2 = 0.0f;
  for (int jj = 0; jj < 32; ++jj) se2 += expf(sc[jj][t] - m2);
  float lse_col = m2 + logf(se2);

  losses[t] = (lse_row - sc[t][t]) + (lse_col - sc[t][t]);
  __syncthreads();

  if (t == 0) {
    float s = 0.0f;
    for (int k = 0; k < 32; ++k) s += losses[k];
    float contrastive = s * (1.0f / 32.0f) * 0.5f;
    float T = tptr[0];
    float l_nn = nn_acc[0] / (T * T) * (1.0f / (32.0f * 32.0f * 50.0f * 1024.0f));
    float lcal = fmaxf(-logf(T), 0.0f);
    out[0] = contrastive + 0.01f * l_nn + 0.1f * lcal * lcal;
  }
}

// ---------------------------------------------------------------------------
// Host launcher. Workspace layout:
//   [0, 4 MiB)        audio bf16   32 x 64 x 1024
//   [4, 68 MiB)       visual bf16  32 x 1024 x 1024
//   [68 MiB, +4 KiB)  clip_sims f32 32 x 32
//   then              nonneg accumulator f32 (zeroed each launch)
// ---------------------------------------------------------------------------
extern "C" void kernel_launch(void* const* d_in, const int* in_sizes, int n_in,
                              void* d_out, int out_size, void* d_ws, size_t ws_size,
                              hipStream_t stream) {
  const float* audio  = (const float*)d_in[0];   // (32, 50, 1024)
  const float* visual = (const float*)d_in[1];   // (32, 1024, 1024)
  const float* temp   = (const float*)d_in[2];   // scalar

  char* ws = (char*)d_ws;
  unsigned short* aB = (unsigned short*)ws;                       // 4 MiB
  unsigned short* vB = (unsigned short*)(ws + (4ull << 20));      // 64 MiB
  float* clip        = (float*)(ws + (68ull << 20));              // 4 KiB
  float* nn          = clip + 1024;

  hipMemsetAsync(nn, 0, sizeof(float), stream);

  norm_rows_kernel<<<32 * 64,   256, 0, stream>>>(audio,  aB,   50,   64);
  norm_rows_kernel<<<32 * 1024, 256, 0, stream>>>(visual, vB, 1024, 1024);
  av_gemm_fused  <<<32 * 32,    256, 0, stream>>>(aB, vB, clip, nn, temp);
  finalize_kernel<<<1, 32, 0, stream>>>(clip, nn, temp, (float*)d_out);
}
